// ArchDecoder_33234456936669
// MI455X (gfx1250) — matrix operand order repaired and checked by reading, not compiled, not run on hardware
//
#include <hip/hip_runtime.h>
#include <hip/hip_bf16.h>

#define V_A   256
#define V_HP  256
#define H_A   2048
#define H_HP  4096
#define DEPTH 12

typedef __attribute__((ext_vector_type(16))) __bf16 v16bf;
typedef __attribute__((ext_vector_type(8)))  __bf16 v8bf;
typedef __attribute__((ext_vector_type(8)))  float  v8f;

// ---------------------------------------------------------------------------
// f32 -> bf16 weight conversion (once per launch; streams 417MB -> 208MB)
// ---------------------------------------------------------------------------
__global__ void f32_to_bf16(const float* __restrict__ src, __bf16* __restrict__ dst, long n) {
    long i = (long)blockIdx.x * blockDim.x + threadIdx.x;
    long stride = (long)gridDim.x * blockDim.x;
    for (; i < n; i += stride) dst[i] = (__bf16)src[i];
}

// ---------------------------------------------------------------------------
// Segmented GEMV via v_wmma_f32_16x16x32_bf16.
//   y[0:M] = sum_over_segments( W_seg[M x K_seg] @ x_seg ) + bias (+ bias2)
// One wave per 16-row strip.  A = 16x32 bf16 weight tile in contiguous
// K order (lanes 0-15: K 0-15, lanes 16-31: K 16-31).  B = x chunk
// replicated across all 16 N columns, so every column of D equals the GEMV
// result; lanes 0/16 extract rows 0-7/8-15 from the f32 accumulator.
// ---------------------------------------------------------------------------
struct GemvSeg {
    const __bf16* W;   // row-major, leading dimension ld (null => unused)
    const __bf16* x;   // bf16 input vector, length >= K
    int K;             // columns to reduce (multiple of 32)
    int ld;            // leading dimension of W
};

__global__ void __launch_bounds__(256)
gemv_bf16_wmma(GemvSeg s0, GemvSeg s1, GemvSeg s2,
               const float* __restrict__ bias, const float* bias2,
               float* __restrict__ y, int M, int relu)
{
    const int lane = threadIdx.x & 31;
    const int wave = threadIdx.x >> 5;
    const int tile = blockIdx.x * (blockDim.x >> 5) + wave;
    const int row0 = tile * 16;
    if (row0 >= M) return;                 // whole wave exits: EXEC all-1 for WMMA

    const int mrow = lane & 15;
    const int hi   = lane >> 4;            // 0: lanes 0-15, 1: lanes 16-31
    const int aoff = hi * 16;              // A K-offset within the 32-chunk
    const int boff = hi * 16;              // B K-offset within the 32-chunk

    v8f acc = {};
    const GemvSeg segs[3] = {s0, s1, s2};
#pragma unroll
    for (int si = 0; si < 3; ++si) {
        const GemvSeg s = segs[si];
        if (!s.W) continue;
        const __bf16* wr = s.W + (size_t)(row0 + mrow) * s.ld + aoff;
        const __bf16* xb = s.x + boff;
        for (int k = 0; k < s.K; k += 32) {
            v8bf alo = *(const v8bf*)(wr + k);        // 16B: K = k+aoff .. +7
            v8bf ahi = *(const v8bf*)(wr + k + 8);    // 16B: K = k+aoff+8 .. +15
            v16bf A = __builtin_shufflevector(alo, ahi,
                        0,1,2,3,4,5,6,7,8,9,10,11,12,13,14,15);
            v16bf B = *(const v16bf*)(xb + k);        // 32B broadcast vec chunk
            acc = __builtin_amdgcn_wmma_f32_16x16x32_bf16(
                      false, A, false, B, (short)0, acc, false, false);
        }
    }

    if (mrow == 0) {                       // lanes 0 and 16 extract the result
        const int base = row0 + hi * 8;
#pragma unroll
        for (int r = 0; r < 8; ++r) {
            float v = acc[r] + bias[base + r];
            if (bias2) v += bias2[base + r];
            if (relu) v = fmaxf(v, 0.0f);
            y[base + r] = v;
        }
    }
}

// ---------------------------------------------------------------------------
// Pointwise LSTM cell update.  Gate order i,f,g,o along 4H.
// c_old[j] = j < split ? c_lo[j] : c_hi[j-split]   (handles the hp concat)
// Also emits bf16 shadow copies of h and c for downstream WMMA GEMVs.
// ---------------------------------------------------------------------------
__device__ __forceinline__ float sigmoidf(float v) { return 1.0f / (1.0f + expf(-v)); }

__global__ void lstm_pointwise(const float* __restrict__ gates,
                               const float* c_lo, const float* c_hi, int split,
                               float* h_out, float* c_out,
                               __bf16* h_bf, __bf16* c_bf, int H)
{
    int j = blockIdx.x * blockDim.x + threadIdx.x;
    if (j >= H) return;
    float ig = gates[j];
    float fg = gates[j + H];
    float gg = gates[j + 2 * H];
    float og = gates[j + 3 * H];
    float c_old = (j < split) ? c_lo[j] : c_hi[j - split];
    float c = sigmoidf(fg) * c_old + sigmoidf(ig) * tanhf(gg);
    float h = sigmoidf(og) * tanhf(c);
    c_out[j] = c;
    h_out[j] = h;
    h_bf[j] = (__bf16)h;
    c_bf[j] = (__bf16)c;
}

// ---------------------------------------------------------------------------
// 256-wide softmax; writes bf16 copy (for next GEMV) and the f32 step output.
// ---------------------------------------------------------------------------
__global__ void softmax256(const float* __restrict__ logits,
                           __bf16* __restrict__ a_bf, float* __restrict__ out)
{
    __shared__ float red[256];
    int t = threadIdx.x;
    float v = logits[t];
    red[t] = v; __syncthreads();
    for (int s = 128; s > 0; s >>= 1) {
        if (t < s) red[t] = fmaxf(red[t], red[t + s]);
        __syncthreads();
    }
    float m = red[0]; __syncthreads();
    float e = expf(v - m);
    red[t] = e; __syncthreads();
    for (int s = 128; s > 0; s >>= 1) {
        if (t < s) red[t] += red[t + s];
        __syncthreads();
    }
    float r = e / red[0];
    a_bf[t] = (__bf16)r;
    out[t] = r;
}

// relu GEMV output (h_sum) is consumed in f32 by lstm_pointwise only.

__global__ void init_state(const float* __restrict__ x_a, const float* __restrict__ x_hp,
                           float* h_a, float* c_a, float* h_hp, float* c_hp,
                           __bf16* h_a_bf, __bf16* h_hp_bf, __bf16* c_hp_bf,
                           __bf16* a_bf, __bf16* ahp_bf)
{
    int i = blockIdx.x * blockDim.x + threadIdx.x;
    if (i < H_A)  { float v = x_a[i];  h_a[i]  = v; c_a[i]  = 0.0f; h_a_bf[i]  = (__bf16)v; }
    if (i < H_HP) { float v = x_hp[i]; h_hp[i] = v; c_hp[i] = 0.0f; h_hp_bf[i] = (__bf16)v;
                    c_hp_bf[i] = (__bf16)0.0f; }
    if (i < V_A)  { a_bf[i] = (__bf16)(1.0f / V_A); ahp_bf[i] = (__bf16)(1.0f / V_HP); }
}

// ---------------------------------------------------------------------------
extern "C" void kernel_launch(void* const* d_in, const int* in_sizes, int n_in,
                              void* d_out, int out_size, void* d_ws, size_t ws_size,
                              hipStream_t stream)
{
    const float* x_a      = (const float*)d_in[0];
    const float* x_hp     = (const float*)d_in[1];
    const float* W_ih_a   = (const float*)d_in[2];
    const float* W_hh_a   = (const float*)d_in[3];
    const float* b_ih_a   = (const float*)d_in[4];
    const float* b_hh_a   = (const float*)d_in[5];
    const float* W_out_a  = (const float*)d_in[6];
    const float* b_out_a  = (const float*)d_in[7];
    const float* W_sum    = (const float*)d_in[8];
    const float* b_sum    = (const float*)d_in[9];
    const float* W_ih_hp  = (const float*)d_in[10];
    const float* W_hh_hp  = (const float*)d_in[11];
    const float* b_ih_hp  = (const float*)d_in[12];
    const float* b_hh_hp  = (const float*)d_in[13];
    const float* W_out_hp = (const float*)d_in[14];
    const float* b_out_hp = (const float*)d_in[15];
    float* out = (float*)d_out;

    // ---- carve workspace: bf16 weight copies + activations ----
    char* ws = (char*)d_ws;
    size_t off = 0;
    auto carve = [&](size_t bytes) -> char* {
        char* p = ws + off;
        off = (off + bytes + 255) & ~(size_t)255;
        return p;
    };
    __bf16* Wb_ih_a   = (__bf16*)carve((size_t)4 * H_A  * V_A        * 2);
    __bf16* Wb_hh_a   = (__bf16*)carve((size_t)4 * H_A  * H_A        * 2);
    __bf16* Wb_out_a  = (__bf16*)carve((size_t)V_A      * H_A        * 2);
    __bf16* Wb_sum    = (__bf16*)carve((size_t)(H_HP/2) * H_HP       * 2);
    __bf16* Wb_ih_hp  = (__bf16*)carve((size_t)4 * H_HP * (V_A+V_HP) * 2);
    __bf16* Wb_hh_hp  = (__bf16*)carve((size_t)4 * H_HP * H_HP       * 2);
    __bf16* Wb_out_hp = (__bf16*)carve((size_t)V_HP     * H_HP       * 2);

    float*  h_a       = (float*)carve(H_A  * 4);
    float*  c_a       = (float*)carve(H_A  * 4);
    float*  gates_a   = (float*)carve(4 * H_A * 4);
    float*  h_hp      = (float*)carve(H_HP * 4);
    float*  c_hp      = (float*)carve(H_HP * 4);
    float*  gates_hp  = (float*)carve(4 * H_HP * 4);
    float*  h_sum     = (float*)carve((H_HP/2) * 4);
    float*  logits_a  = (float*)carve(V_A  * 4);
    float*  logits_hp = (float*)carve(V_HP * 4);

    __bf16* h_a_bf    = (__bf16*)carve(H_A  * 2);
    __bf16* c_a_bf    = (__bf16*)carve(H_A  * 2);   // written, unused (generic kernel)
    __bf16* h_hp_bf   = (__bf16*)carve(H_HP * 2);
    __bf16* c_hp_bf   = (__bf16*)carve(H_HP * 2);
    __bf16* a_bf      = (__bf16*)carve(V_A  * 2);
    __bf16* ahp_bf    = (__bf16*)carve(V_HP * 2);

    auto conv = [&](const float* s, __bf16* d, long n) {
        int blocks = (int)((n + 2047) / 2048);
        if (blocks > 8192) blocks = 8192;
        f32_to_bf16<<<blocks, 256, 0, stream>>>(s, d, n);
    };
    conv(W_ih_a,   Wb_ih_a,   (long)4 * H_A  * V_A);
    conv(W_hh_a,   Wb_hh_a,   (long)4 * H_A  * H_A);
    conv(W_out_a,  Wb_out_a,  (long)V_A      * H_A);
    conv(W_sum,    Wb_sum,    (long)(H_HP/2) * H_HP);
    conv(W_ih_hp,  Wb_ih_hp,  (long)4 * H_HP * (V_A + V_HP));
    conv(W_hh_hp,  Wb_hh_hp,  (long)4 * H_HP * H_HP);
    conv(W_out_hp, Wb_out_hp, (long)V_HP     * H_HP);

    init_state<<<H_HP / 256, 256, 0, stream>>>(x_a, x_hp, h_a, c_a, h_hp, c_hp,
                                               h_a_bf, h_hp_bf, c_hp_bf, a_bf, ahp_bf);

    const GemvSeg nil = {nullptr, nullptr, 0, 0};
    auto gemv = [&](GemvSeg a, GemvSeg b, GemvSeg c,
                    const float* bias, const float* bias2,
                    float* y, int M, int relu) {
        int blocks = (M / 16 + 7) / 8;     // 8 waves (16-row strips) per block
        gemv_bf16_wmma<<<blocks, 256, 0, stream>>>(a, b, c, bias, bias2, y, M, relu);
    };

    for (int s = 0; s < DEPTH; ++s) {
        // --- arch LSTM: gates = W_ih@a + W_hh@h_a + b_ih + b_hh (one kernel) ---
        gemv({Wb_ih_a, a_bf, V_A, V_A}, {Wb_hh_a, h_a_bf, H_A, H_A}, nil,
             b_ih_a, b_hh_a, gates_a, 4 * H_A, 0);
        lstm_pointwise<<<H_A / 256, 256, 0, stream>>>(gates_a, c_a, c_a, H_A,
                                                      h_a, c_a, h_a_bf, c_a_bf, H_A);

        // --- arch out + softmax -> a (also step output) ---
        gemv({Wb_out_a, h_a_bf, H_A, H_A}, nil, nil,
             b_out_a, nullptr, logits_a, V_A, 0);
        softmax256<<<1, 256, 0, stream>>>(logits_a, a_bf, out + s * V_A);

        // --- summarizer: h_sum = relu(W_sum @ h_hp(prev) + b_sum) ---
        gemv({Wb_sum, h_hp_bf, H_HP, H_HP}, nil, nil,
             b_sum, nullptr, h_sum, H_HP / 2, 1);

        // --- hp LSTM (swapped hidden/cell): gates = W_ih@[a|a_hp] + W_hh@c_hp + b ---
        gemv({Wb_ih_hp, a_bf, V_A, V_A + V_HP},
             {Wb_ih_hp + V_A, ahp_bf, V_HP, V_A + V_HP},
             {Wb_hh_hp, c_hp_bf, H_HP, H_HP},
             b_ih_hp, b_hh_hp, gates_hp, 4 * H_HP, 0);
        lstm_pointwise<<<H_HP / 256, 256, 0, stream>>>(gates_hp, h_a, h_sum, H_A,
                                                       h_hp, c_hp, h_hp_bf, c_hp_bf, H_HP);

        // --- hp out + softmax -> a_hp (also step output) ---
        gemv({Wb_out_hp, h_hp_bf, H_HP, H_HP}, nil, nil,
             b_out_hp, nullptr, logits_hp, V_HP, 0);
        softmax256<<<1, 256, 0, stream>>>(logits_hp, ahp_bf, out + DEPTH * V_A + s * V_HP);
    }

    (void)in_sizes; (void)n_in; (void)out_size; (void)ws_size;
}
